// RoIBBox_62440234549453
// MI455X (gfx1250) — compile-verified, matrix operand order checked by reading, not run on hardware
//
#include <hip/hip_runtime.h>
#include <math.h>

// ---------------------------------------------------------------------------
// RoI bbox pipeline for MI455X (gfx1250, wave32).
//  K1 decode:  elementwise delta decode, f32.
//  K2 nms:     1 block / batch image, 1024 threads, scores resident in LDS
//              (147KB < 320KB WGP LDS), boxes streamed from L2-resident global.
//              300 fused suppress+argmax passes (exact greedy NMS semantics).
//  K3 select:  1 wave / batch. IoU(300x64) + stable-sort ranks computed as a
//              0/1-indicator row-sum on the matrix core:
//              rank = C * ones  via  v_wmma_f32_16x16x32_f16 (exact in f16/f32).
//              A-matrix built branchlessly from ds_load_b128 reads.
// ---------------------------------------------------------------------------

typedef __attribute__((ext_vector_type(16))) _Float16 v16h;
typedef __attribute__((ext_vector_type(8)))  float    v8f;

#define BATCH   16
#define NANCH   36864      // 64*64*9
#define TOPN    300
#define NPAD    320        // TOPN padded to K-chunk multiple (32)
#define NGT     64
#define NPOS    64
#define NROI    128        // NPOS + 64 zeros
#define NEGINF  (-1e30f)

// ---------------------------------------------------------------- K1: decode
__global__ __launch_bounds__(256)
void decode_kernel(const float* __restrict__ deltas,
                   const float* __restrict__ anchors,
                   float* __restrict__ boxes) {
    int i = blockIdx.x * blockDim.x + threadIdx.x;
    if (i >= BATCH * NANCH) return;
    float4 a = ((const float4*)anchors)[i];
    float4 d = ((const float4*)deltas)[i];
    float ah  = a.z - a.x;
    float aw  = a.w - a.y;
    float acy = a.x + 0.5f * ah;
    float acx = a.y + 0.5f * aw;
    float h   = expf(d.z) * ah;
    float w   = expf(d.w) * aw;
    float cy  = d.x * ah + acy;
    float cx  = d.y * aw + acx;
    float y1  = cy - 0.5f * h;
    float x1  = cx - 0.5f * w;
    ((float4*)boxes)[i] = make_float4(y1, x1, y1 + h, x1 + w);
}

// ------------------------------------------------------------------- K2: NMS
__global__ __launch_bounds__(1024)
void nms_kernel(const float* __restrict__ scores,
                const float* __restrict__ boxes,
                float* __restrict__ nms_out /* BATCH*TOPN*4, clipped/zeroed */) {
    __shared__ float  s_sc[NANCH];          // 147456 B score table
    __shared__ float  s_rv[32];
    __shared__ int    s_ri[32];
    __shared__ float4 s_selbox;             // raw (unclipped) selected box
    __shared__ int    s_selidx;

    const int b   = blockIdx.x;
    const int tid = threadIdx.x;
    const float*  sc = scores + (size_t)b * NANCH;
    const float4* bx = (const float4*)boxes + (size_t)b * NANCH;
    float4* onms = (float4*)nms_out + (size_t)b * TOPN;

    for (int s = tid; s < NANCH; s += 1024) s_sc[s] = sc[s];
    __syncthreads();

    int    prevIdx = -1;
    float4 pb      = make_float4(0.f, 0.f, 0.f, 0.f);

    for (int t = 0; t < TOPN; ++t) {
        // fused: suppress by previous pick, then argmax (tie -> smallest idx)
        float best = -INFINITY;
        int   bidx = 0x7fffffff;
        float pa = (pb.z - pb.x) * (pb.w - pb.y);
        for (int s = tid; s < NANCH; s += 1024) {
            float v = s_sc[s];
            if (t > 0 && v > NEGINF * 0.5f) {        // skip already-suppressed
                if (s == prevIdx) {
                    v = NEGINF; s_sc[s] = v;
                } else {
                    float4 q  = bx[s];
                    float yy1 = fmaxf(pb.x, q.x), xx1 = fmaxf(pb.y, q.y);
                    float yy2 = fminf(pb.z, q.z), xx2 = fminf(pb.w, q.w);
                    float inter = fmaxf(yy2 - yy1, 0.f) * fmaxf(xx2 - xx1, 0.f);
                    float qa  = (q.z - q.x) * (q.w - q.y);
                    float iou = inter / (pa + qa - inter + 1e-7f);
                    if (iou >= 0.5f) { v = NEGINF; s_sc[s] = v; }
                }
            }
            if (v > best || (v == best && s < bidx)) { best = v; bidx = s; }
        }
        // wave32 reduction
        for (int off = 16; off; off >>= 1) {
            float ov = __shfl_down(best, off, 32);
            int   oi = __shfl_down(bidx, off, 32);
            if (ov > best || (ov == best && oi < bidx)) { best = ov; bidx = oi; }
        }
        if ((tid & 31) == 0) { s_rv[tid >> 5] = best; s_ri[tid >> 5] = bidx; }
        __syncthreads();
        if (tid < 32) {
            best = s_rv[tid]; bidx = s_ri[tid];
            for (int off = 16; off; off >>= 1) {
                float ov = __shfl_down(best, off, 32);
                int   oi = __shfl_down(bidx, off, 32);
                if (ov > best || (ov == best && oi < bidx)) { best = ov; bidx = oi; }
            }
            if (tid == 0) {
                float4 q = bx[bidx];
                s_selbox = q;
                s_selidx = bidx;
                bool valid = best > NEGINF * 0.5f;
                float4 o;
                o.x = valid ? fminf(fmaxf(q.x, 0.f), 1.f) : 0.f;
                o.y = valid ? fminf(fmaxf(q.y, 0.f), 1.f) : 0.f;
                o.z = valid ? fminf(fmaxf(q.z, 0.f), 1.f) : 0.f;
                o.w = valid ? fminf(fmaxf(q.w, 0.f), 1.f) : 0.f;
                onms[t] = o;
            }
        }
        __syncthreads();
        prevIdx = s_selidx;
        pb      = s_selbox;
    }
}

// ------------------------------------------- K3: select_rois (WMMA rank sums)
__global__ __launch_bounds__(32)
void select_kernel(const float* __restrict__ nms_boxes,
                   const float* __restrict__ gt,
                   float* __restrict__ out_roi,   // BATCH*NROI*4
                   float* __restrict__ out_idx) { // BATCH*NPOS
    __shared__ float4 s_nb[TOPN];
    __shared__ float4 s_gt[NGT];
    __shared__ __align__(16) float s_m[NPAD];   // merged, padded with -INF
    __shared__ int    s_gb[TOPN];               // argmax gt (first max)
    __shared__ int    s_rank[304];              // 19*16 rows

    const int b    = blockIdx.x;
    const int lane = threadIdx.x;
    const float4* nb = (const float4*)nms_boxes + (size_t)b * TOPN;
    const float4* g  = (const float4*)gt + (size_t)b * NGT;

    for (int i = lane; i < TOPN; i += 32) s_nb[i] = nb[i];
    for (int i = lane; i < NGT;  i += 32) s_gt[i] = g[i];
    // pad merged so out-of-range columns contribute 0 to indicator sums
    for (int i = TOPN + lane; i < NPAD; i += 32) s_m[i] = -INFINITY;
    __syncthreads();

    // per nms-box: max IoU / first-argmax over 64 gt boxes (merged in [0,1])
    for (int i = lane; i < TOPN; i += 32) {
        float4 a  = s_nb[i];
        float  aa = (a.z - a.x) * (a.w - a.y);
        float bestv = -INFINITY; int bj = 0;
        for (int j = 0; j < NGT; ++j) {
            float4 q  = s_gt[j];
            float yy1 = fmaxf(a.x, q.x), xx1 = fmaxf(a.y, q.y);
            float yy2 = fminf(a.z, q.z), xx2 = fminf(a.w, q.w);
            float inter = fmaxf(yy2 - yy1, 0.f) * fmaxf(xx2 - xx1, 0.f);
            float qa  = (q.z - q.x) * (q.w - q.y);
            float iou = inter / (aa + qa - inter + 1e-7f);
            if (iou > bestv) { bestv = iou; bj = j; }
        }
        s_m[i] = bestv; s_gb[i] = bj;
    }
    __syncthreads();

    // Stable-descending-sort rank via matrix core:
    // rank_i = sum_j [ m_j > m_i  ||  (m_j == m_i && j < i) ]
    // 0/1 indicators packed as f16 A (16x32 tile), B = ones => D rows = sums.
    v16h ones;
    #pragma unroll
    for (int e = 0; e < 16; ++e) ones[e] = (_Float16)1.0f;
    const _Float16 h1 = (_Float16)1.0f;
    const _Float16 h0 = (_Float16)0.0f;

    const int m16 = lane & 15;
    const int hi  = lane >> 4;

    for (int tile = 0; tile < 19; ++tile) {          // ceil(300/16)
        const int   row = tile * 16 + m16;           // rows >= 300: ignored
        const float mi  = s_m[row < NPAD ? row : 0]; // row <= 303 < 320 always
        v8f acc = {};
        for (int ch = 0; ch < 10; ++ch) {            // 320/32 K-chunks
            // 16-bit A-matrix 16x32 layout: lane L%16 = row; lane L/16 picks
            // K runs [hi*8, hi*8+8) and [16+hi*8, 16+hi*8+8) of this chunk.
            const int base = ch * 32 + hi * 8;
            float4 g0 = *(const float4*)&s_m[base + 0];
            float4 g1 = *(const float4*)&s_m[base + 4];
            float4 g2 = *(const float4*)&s_m[base + 16];
            float4 g3 = *(const float4*)&s_m[base + 20];
            float mj[16] = { g0.x, g0.y, g0.z, g0.w,  g1.x, g1.y, g1.z, g1.w,
                             g2.x, g2.y, g2.z, g2.w,  g3.x, g3.y, g3.z, g3.w };
            v16h amat;
            #pragma unroll
            for (int e = 0; e < 16; ++e) {
                const int j = base + ((e < 8) ? e : (e + 8)); // second run: +16
                const float m = mj[e];
                // branchless 0/1 indicator (padded m = -INF -> always 0)
                const bool beats = (m > mi) | ((m == mi) & (j < row));
                amat[e] = beats ? h1 : h0;
            }
            acc = __builtin_amdgcn_wmma_f32_16x16x32_f16(
                      false, amat, false, ones, (short)0, acc, false, false);
        }
        // D layout: VGPR r holds M = r + 8*(lane/16), N = lane%16 (all N equal)
        if (m16 == 0) {
            #pragma unroll
            for (int r = 0; r < 8; ++r)
                s_rank[tile * 16 + hi * 8 + r] = (int)(acc[r] + 0.5f);
        }
    }
    __syncthreads();

    // ranks over real rows form a permutation of 0..299 -> conflict-free
    float4* roi  = (float4*)out_roi + (size_t)b * NROI;
    float*  oidx = out_idx + (size_t)b * NPOS;
    for (int i = lane; i < TOPN; i += 32) {
        int rk = s_rank[i];
        if (rk < NPOS) {
            roi[rk]  = s_nb[i];
            oidx[rk] = (float)s_gb[i];
        }
    }
    for (int i = lane; i < NROI - NPOS; i += 32)
        roi[NPOS + i] = make_float4(0.f, 0.f, 0.f, 0.f);
}

// ------------------------------------------------------------------- launcher
extern "C" void kernel_launch(void* const* d_in, const int* in_sizes, int n_in,
                              void* d_out, int out_size, void* d_ws, size_t ws_size,
                              hipStream_t stream) {
    const float* deltas  = (const float*)d_in[0];  // (B,64,64,36) == (B,N,4)
    const float* labels  = (const float*)d_in[1];  // (B,64,64,9)  == (B,N)
    const float* anchors = (const float*)d_in[2];  // (B,N,4)
    const float* gt      = (const float*)d_in[3];  // (B,64,4)

    float* boxes = (float*)d_ws;                            // B*N*4 f32 (9.4MB)
    float* nmsb  = boxes + (size_t)BATCH * NANCH * 4;       // B*TOPN*4

    float* out_roi = (float*)d_out;                         // B*128*4
    float* out_idx = out_roi + (size_t)BATCH * NROI * 4;    // B*64

    decode_kernel<<<(BATCH * NANCH + 255) / 256, 256, 0, stream>>>(deltas, anchors, boxes);
    nms_kernel<<<BATCH, 1024, 0, stream>>>(labels, boxes, nmsb);
    select_kernel<<<BATCH, 32, 0, stream>>>(nmsb, gt, out_roi, out_idx);
}
